// MinGRU_10368051052871
// MI455X (gfx1250) — matrix operand order repaired and checked by reading, not compiled, not run on hardware
//
#include <hip/hip_runtime.h>
#include <stdint.h>

typedef __attribute__((ext_vector_type(16))) __bf16 v16bf;
typedef __attribute__((ext_vector_type(8)))  __bf16 v8bf;
typedef __attribute__((ext_vector_type(4)))  __bf16 v4bf;
typedef __attribute__((ext_vector_type(8)))  float  v8f;

#define T_SEQ   8192
#define BATCH   8
#define HDIM    512
#define NDIM    1536                 // 3*H
#define KDIM    512
#define M_TOTAL (BATCH * T_SEQ)      // 65536

// ---- GEMM tiling ----
#define BM 128
#define BN 128
#define BK 32
#define KS 40                        // padded LDS k-stride (80B rows -> conflict-free b128 reads)
#define NK (KDIM / BK)               // 16 k-steps

// ---- scan chunking ----
#define TC  128
#define NCH (T_SEQ / TC)             // 64 chunks

union V16U { v16bf v; v8bf h[2]; };

__device__ __forceinline__ v8f wmma_bf16(v16bf a, v16bf b, v8f c) {
    // D = A x B + C, 16x16x32 bf16 -> f32
    return __builtin_amdgcn_wmma_f32_16x16x32_bf16(
        /*neg_a=*/false, a, /*neg_b=*/false, b,
        /*c_mod=*/(short)0, c, /*reuse_a=*/false, /*reuse_b=*/false);
}

__device__ __forceinline__ void cvt_hi_lo(float x, __bf16 &hi, __bf16 &lo) {
    hi = (__bf16)x;
    lo = (__bf16)(x - (float)hi);
}

// Split fp32 -> (bf16_hi, bf16_lo) arrays, 4 elements/thread (b128 in, 2x b64 out).
__global__ __launch_bounds__(256) void mingru_cvt_split(
    const float* __restrict__ src, __bf16* __restrict__ hi, __bf16* __restrict__ lo, int n4)
{
    const int i = blockIdx.x * 256 + threadIdx.x;
    if (i >= n4) return;
    const float4 v = ((const float4*)src)[i];
    const float e[4] = {v.x, v.y, v.z, v.w};
    v4bf h, l;
#pragma unroll
    for (int j = 0; j < 4; ++j) { __bf16 hh, ll; cvt_hi_lo(e[j], hh, ll); h[j] = hh; l[j] = ll; }
    ((v4bf*)hi)[i] = h;
    ((v4bf*)lo)[i] = l;
}

// proj = A[M,K] * W[N,K]^T with bf16x3 split accumulation.
// 128x128x32 block tile, 8 waves of 32(M)x64(N) -> 24 WMMA per 24 LDS b128 reads per wave.
// Tiles staged LDS-direct via GLOBAL_LOAD_ASYNC_TO_LDS_B128 (ASYNCcnt), double-buffered.
__global__ __launch_bounds__(256) void mingru_gemm_bf16x3(
    const __bf16* __restrict__ Ahi, const __bf16* __restrict__ Alo,   // [M_TOTAL][KDIM]
    const __bf16* __restrict__ Whi, const __bf16* __restrict__ Wlo,   // [NDIM][KDIM]
    float* __restrict__ C)                                            // [M_TOTAL][NDIM]
{
    __shared__ __attribute__((aligned(16))) __bf16 As_hi[2][BM][KS];
    __shared__ __attribute__((aligned(16))) __bf16 As_lo[2][BM][KS];
    __shared__ __attribute__((aligned(16))) __bf16 Bs_hi[2][BN][KS];
    __shared__ __attribute__((aligned(16))) __bf16 Bs_lo[2][BN][KS];

    const int tid    = threadIdx.x;
    const int lane   = tid & 31;
    const int wid    = tid >> 5;
    const int wave_m = wid >> 1;     // 0..3 -> 32-row strip
    const int wave_n = wid & 1;      // 0..1 -> 64-col strip
    const int l15    = lane & 15;
    const int lhalf  = lane >> 4;

    const int block_n = blockIdx.x;  // 0..11
    const int block_m = blockIdx.y;  // 0..511

    const __bf16* AhiBlk = Ahi + (size_t)block_m * BM * KDIM;
    const __bf16* AloBlk = Alo + (size_t)block_m * BM * KDIM;
    const __bf16* WhiBlk = Whi + (size_t)block_n * BN * KDIM;
    const __bf16* WloBlk = Wlo + (size_t)block_n * BN * KDIM;

    // --- per-thread async-copy assignments ---
    // Each tile (A or B, hi or lo): 128 rows x 32 bf16 = 512 b128 transfers -> 2 per thread.
    const int f0 = tid, f1 = tid + 256;
    const int r0 = f0 >> 2, c0 = (f0 & 3) * 8;
    const int r1 = f1 >> 2, c1 = (f1 & 3) * 8;

    // Running global source addresses (bumped +64B per k-step)
    unsigned long long gAh0 = (unsigned long long)(uintptr_t)(AhiBlk + (size_t)r0 * KDIM + c0);
    unsigned long long gAh1 = (unsigned long long)(uintptr_t)(AhiBlk + (size_t)r1 * KDIM + c1);
    unsigned long long gAl0 = (unsigned long long)(uintptr_t)(AloBlk + (size_t)r0 * KDIM + c0);
    unsigned long long gAl1 = (unsigned long long)(uintptr_t)(AloBlk + (size_t)r1 * KDIM + c1);
    unsigned long long gBh0 = (unsigned long long)(uintptr_t)(WhiBlk + (size_t)r0 * KDIM + c0);
    unsigned long long gBh1 = (unsigned long long)(uintptr_t)(WhiBlk + (size_t)r1 * KDIM + c1);
    unsigned long long gBl0 = (unsigned long long)(uintptr_t)(WloBlk + (size_t)r0 * KDIM + c0);
    unsigned long long gBl1 = (unsigned long long)(uintptr_t)(WloBlk + (size_t)r1 * KDIM + c1);

    // LDS byte offsets (low 32 bits of generic pointer to __shared__ = LDS address)
    uint32_t lAh0[2], lAh1[2], lAl0[2], lAl1[2], lBh0[2], lBh1[2], lBl0[2], lBl1[2];
#pragma unroll
    for (int b = 0; b < 2; ++b) {
        lAh0[b] = (uint32_t)(uintptr_t)&As_hi[b][r0][c0];
        lAh1[b] = (uint32_t)(uintptr_t)&As_hi[b][r1][c1];
        lAl0[b] = (uint32_t)(uintptr_t)&As_lo[b][r0][c0];
        lAl1[b] = (uint32_t)(uintptr_t)&As_lo[b][r1][c1];
        lBh0[b] = (uint32_t)(uintptr_t)&Bs_hi[b][r0][c0];
        lBh1[b] = (uint32_t)(uintptr_t)&Bs_hi[b][r1][c1];
        lBl0[b] = (uint32_t)(uintptr_t)&Bs_lo[b][r0][c0];
        lBl1[b] = (uint32_t)(uintptr_t)&Bs_lo[b][r1][c1];
    }

#define ASYNC_CP(lds, gaddr) \
    asm volatile("global_load_async_to_lds_b128 %0, %1, off" :: "v"(lds), "v"(gaddr) : "memory")

    // Issue 8 async b128 copies into buffer `buf`, then advance one k-step (64 bytes).
    auto stageAsyncAdvance = [&](int buf) {
        ASYNC_CP(lAh0[buf], gAh0);  ASYNC_CP(lAh1[buf], gAh1);
        ASYNC_CP(lAl0[buf], gAl0);  ASYNC_CP(lAl1[buf], gAl1);
        ASYNC_CP(lBh0[buf], gBh0);  ASYNC_CP(lBh1[buf], gBh1);
        ASYNC_CP(lBl0[buf], gBl0);  ASYNC_CP(lBl1[buf], gBl1);
        gAh0 += BK * 2; gAh1 += BK * 2; gAl0 += BK * 2; gAl1 += BK * 2;
        gBh0 += BK * 2; gBh1 += BK * 2; gBl0 += BK * 2; gBl1 += BK * 2;
    };

    v8f acc[2][4] = {};

    stageAsyncAdvance(0);
    asm volatile("s_wait_asynccnt 0x0" ::: "memory");
    __syncthreads();

#pragma unroll 1                      // keep the loop rolled: tight body, low VGPR pressure
    for (int ks = 0; ks < NK; ++ks) {
        const int cur = ks & 1;
        if (ks + 1 < NK) stageAsyncAdvance(cur ^ 1);   // overlap async copies with WMMA

        // A fragments: two 16-row tiles; lane l15 row, halves at k = lhalf*8 and +16
        V16U ah[2], al[2];
#pragma unroll
        for (int r = 0; r < 2; ++r) {
            const int arow = wave_m * 32 + r * 16 + l15;
            const int ko   = lhalf * 8;
            ah[r].h[0] = *(v8bf*)&As_hi[cur][arow][ko];
            ah[r].h[1] = *(v8bf*)&As_hi[cur][arow][ko + 16];
            al[r].h[0] = *(v8bf*)&As_lo[cur][arow][ko];
            al[r].h[1] = *(v8bf*)&As_lo[cur][arow][ko + 16];
        }

#pragma unroll
        for (int t = 0; t < 4; ++t) {
            // B fragment: lane n holds W row (col), halves at k = lhalf*8 and +16
            const int bcol = wave_n * 64 + t * 16 + l15;
            const int k0   = lhalf * 8;
            V16U bh, bl;
            bh.h[0] = *(v8bf*)&Bs_hi[cur][bcol][k0];
            bh.h[1] = *(v8bf*)&Bs_hi[cur][bcol][k0 + 16];
            bl.h[0] = *(v8bf*)&Bs_lo[cur][bcol][k0];
            bl.h[1] = *(v8bf*)&Bs_lo[cur][bcol][k0 + 16];
#pragma unroll
            for (int r = 0; r < 2; ++r) {
                acc[r][t] = wmma_bf16(ah[r].v, bh.v, acc[r][t]);   // hi*hi
                acc[r][t] = wmma_bf16(ah[r].v, bl.v, acc[r][t]);   // hi*lo
                acc[r][t] = wmma_bf16(al[r].v, bh.v, acc[r][t]);   // lo*hi
            }
        }

        asm volatile("s_wait_asynccnt 0x0" ::: "memory");  // drain copies into next buffer
        __syncthreads();
    }

    // Epilogue: C/D layout -> VGPR j, lanes 0-15 = row j, lanes 16-31 = row j+8
    const int colBase = block_n * BN + wave_n * 64 + l15;
#pragma unroll
    for (int r = 0; r < 2; ++r) {
        const size_t rowBase = (size_t)block_m * BM + wave_m * 32 + r * 16 + lhalf * 8;
#pragma unroll
        for (int t = 0; t < 4; ++t) {
            const int col = colBase + t * 16;
#pragma unroll
            for (int j = 0; j < 8; ++j)
                C[(rowBase + j) * (size_t)NDIM + col] = acc[r][t][j];
        }
    }
#undef ASYNC_CP
}

// out_t = sigmoid(-gate)*out_{t-1} + sigmoid(gate)*g(inner)  (linear-space MinGRU recurrence)
__device__ __forceinline__ void coeff_val(float gate, float inner, float &cf, float &vl) {
    float sg;
    if (gate >= 0.f) { float e = __expf(-gate); float d = 1.f / (1.f + e); sg = d;     cf = e * d; }
    else             { float e = __expf(gate);  float d = 1.f / (1.f + e); sg = e * d; cf = d;     }
    float gi;
    if (inner >= 0.f) gi = inner + 0.5f;
    else              { float e = __expf(inner); gi = e / (1.f + e); }
    vl = sg * gi;
}

// Pass A: per-chunk composition (P = prod coeff, S = chunk result from state 0)
__global__ __launch_bounds__(256) void mingru_scanA(
    const float* __restrict__ proj, float* __restrict__ Pc, float* __restrict__ Sc)
{
    const int h = blockIdx.x * 256 + threadIdx.x;
    const int chunk = blockIdx.y;
    const int b = blockIdx.z;
    float P = 1.f, S = 0.f;
    const int t0 = chunk * TC;
    for (int t = t0; t < t0 + TC; ++t) {
        const size_t base = (size_t)(b * T_SEQ + t) * NDIM;
        float inner = proj[base + h];
        float gate  = proj[base + HDIM + h];
        float cf, vl; coeff_val(gate, inner, cf, vl);
        P *= cf;
        S  = cf * S + vl;
    }
    const size_t idx = (size_t)(b * NCH + chunk) * HDIM + h;
    Pc[idx] = P; Sc[idx] = S;
}

// Pass B: exclusive scan of chunk partials -> per-chunk carry-in state
__global__ __launch_bounds__(256) void mingru_scanB(
    const float* __restrict__ Pc, const float* __restrict__ Sc, float* __restrict__ carry)
{
    const int gid = blockIdx.x * 256 + threadIdx.x;      // 0..4095
    const int b = gid >> 9;
    const int h = gid & (HDIM - 1);
    float s = 0.f;
    for (int c = 0; c < NCH; ++c) {
        const size_t idx = (size_t)(b * NCH + c) * HDIM + h;
        carry[idx] = s;
        s = Pc[idx] * s + Sc[idx];
    }
}

// Pass C: replay chunk with carry-in, apply highway gate, write new hidden
__global__ __launch_bounds__(256) void mingru_scanC(
    const float* __restrict__ proj, const float* __restrict__ carry,
    const float* __restrict__ h_in, float* __restrict__ h_out)
{
    const int h = blockIdx.x * 256 + threadIdx.x;
    const int chunk = blockIdx.y;
    const int b = blockIdx.z;
    float s = carry[(size_t)(b * NCH + chunk) * HDIM + h];
    const int t0 = chunk * TC;
    for (int t = t0; t < t0 + TC; ++t) {
        const size_t base = (size_t)(b * T_SEQ + t) * NDIM;
        float inner = proj[base + h];
        float gate  = proj[base + HDIM + h];
        float hw    = proj[base + 2 * HDIM + h];
        float cf, vl; coeff_val(gate, inner, cf, vl);
        s = cf * s + vl;
        float g;
        if (hw >= 0.f) { float e = __expf(-hw); g = 1.f / (1.f + e); }
        else           { float e = __expf(hw);  g = e / (1.f + e);   }
        const size_t hidx = (size_t)(b * T_SEQ + t) * HDIM + h;
        const float hp = h_in[hidx];
        h_out[hidx] = g * s + (1.f - g) * hp;
    }
}

extern "C" void kernel_launch(void* const* d_in, const int* in_sizes, int n_in,
                              void* d_out, int out_size, void* d_ws, size_t ws_size,
                              hipStream_t stream) {
    const float* hidden = (const float*)d_in[0];   // [8,8192,512] f32
    const float* Ws     = (const float*)d_in[1];   // [2,1536,512] f32
    float* out = (float*)d_out;                    // [8,8192,512] f32

    // workspace layout (all 16B-aligned slices)
    float* proj = (float*)d_ws;                                    // M*N f32 (384 MB)
    const size_t projElems = (size_t)M_TOTAL * NDIM;
    const size_t partElems = (size_t)BATCH * NCH * HDIM;           // 262144
    float* Pc    = proj + projElems;
    float* Sc    = Pc   + partElems;
    float* carry = Sc   + partElems;
    __bf16* Ahi = (__bf16*)(carry + partElems);                    // [M][K] bf16 (64 MB)
    __bf16* Alo = Ahi + (size_t)M_TOTAL * KDIM;
    __bf16* Whi = Alo + (size_t)M_TOTAL * KDIM;                    // [N][K] bf16 (1.5 MB)
    __bf16* Wlo = Whi + (size_t)NDIM * KDIM;

    const int a4 = (M_TOTAL * KDIM) / 4;   // 8,388,608 float4s
    const int w4 = (NDIM * KDIM) / 4;      // 196,608 float4s

    for (int l = 0; l < 2; ++l) {
        const float* Ain = (l == 0) ? hidden : out;
        const float* Wl  = Ws + (size_t)l * NDIM * KDIM;
        mingru_cvt_split<<<dim3((a4 + 255) / 256), 256, 0, stream>>>(Ain, Ahi, Alo, a4);
        mingru_cvt_split<<<dim3((w4 + 255) / 256), 256, 0, stream>>>(Wl, Whi, Wlo, w4);
        mingru_gemm_bf16x3<<<dim3(NDIM / BN, M_TOTAL / BM), 256, 0, stream>>>(Ahi, Alo, Whi, Wlo, proj);
        mingru_scanA<<<dim3(HDIM / 256, NCH, BATCH), 256, 0, stream>>>(proj, Pc, Sc);
        mingru_scanB<<<dim3((BATCH * HDIM) / 256), 256, 0, stream>>>(Pc, Sc, carry);
        mingru_scanC<<<dim3(HDIM / 256, NCH, BATCH), 256, 0, stream>>>(proj, carry, Ain, out);
    }
}
